// MOARecRobertaEncoder_67130338836513
// MI455X (gfx1250) — matrix operand order repaired and checked by reading, not compiled, not run on hardware
//
#include <hip/hip_runtime.h>

// ---------------- dims ----------------
constexpr int Bb  = 8;
constexpr int Ss  = 512;
constexpr int Hh  = 768;
constexpr int NHh = 12;
constexpr int FFf = 3072;
constexpr int Ee  = 8;
constexpr int Rr  = 64;
constexpr int NLl = 4;
constexpr int DHh = Hh / NHh;   // 64
constexpr int ERr = Ee * Rr;    // 512
constexpr int Mm  = Bb * Ss;    // 4096

// ---------------- WMMA types ----------------
typedef __attribute__((ext_vector_type(16))) __bf16 v16bf;
typedef __attribute__((ext_vector_type(8)))  float  v8f;

union FragU { v16bf v; uint4 q[2]; };

__device__ __forceinline__ __bf16 f2bf(float f) {
  union { float f; unsigned u; } x; x.f = f;
  unsigned r = x.u + 0x7FFFu + ((x.u >> 16) & 1u);
  unsigned short h = (unsigned short)(r >> 16);
  union { unsigned short s; __bf16 b; } y; y.s = h;
  return y.b;
}

__device__ __forceinline__ float gelu_exact(float x) {
  return 0.5f * x * (1.0f + erff(x * 0.70710678118654752440f));
}

__device__ __forceinline__ float wave_sum(float v) {
  #pragma unroll
  for (int o = 16; o > 0; o >>= 1) v += __shfl_xor(v, o, 32);
  return v;
}
__device__ __forceinline__ float wave_max(float v) {
  #pragma unroll
  for (int o = 16; o > 0; o >>= 1) v = fmaxf(v, __shfl_xor(v, o, 32));
  return v;
}

// A fragment (16x32, row-major source [rows, K], ld in elements):
// lanes 0-15: row m, K in [0..7] and [16..23]; lanes 16-31: row m, K in [8..15] and [24..31]
__device__ __forceinline__ v16bf load_frag_A(const __bf16* base, int ld) {
  int lane = threadIdx.x & 31;
  int half = lane >> 4;
  int m    = lane & 15;
  const __bf16* p = base + m * ld + half * 8;
  FragU f;
  f.q[0] = *reinterpret_cast<const uint4*>(p);
  f.q[1] = *reinterpret_cast<const uint4*>(p + 16);
  return f.v;
}

// B fragment (32x16 KxN), sourced from B^T row-major [N, K]:
// lanes 0-15: col n, K in [0..7] and [16..23]; lanes 16-31: col n, K in [8..15] and [24..31]
__device__ __forceinline__ v16bf load_frag_B(const __bf16* base, int ld) {
  int lane = threadIdx.x & 31;
  int half = lane >> 4;
  int n    = lane & 15;
  const __bf16* p = base + n * ld + half * 8;
  FragU f;
  f.q[0] = *reinterpret_cast<const uint4*>(p);
  f.q[1] = *reinterpret_cast<const uint4*>(p + 16);
  return f.v;
}

__device__ __forceinline__ v8f wmma_bf16(v16bf a, v16bf b, v8f c) {
  return __builtin_amdgcn_wmma_f32_16x16x32_bf16(false, a, false, b, (short)0, c,
                                                 false, false);
}

// ---------------- convert helpers ----------------
__global__ __launch_bounds__(256) void cvt_bf16_kernel(const float* __restrict__ x,
                                                       __bf16* __restrict__ y, int n) {
  int i = blockIdx.x * 256 + threadIdx.x;
  if (i < n) y[i] = f2bf(x[i]);
}

// W [K, N] fp32 row-major  ->  WT [N, K] bf16 row-major
__global__ __launch_bounds__(256) void transpose_cvt_kernel(const float* __restrict__ W,
                                                            __bf16* __restrict__ WT,
                                                            int K, int N) {
  __shared__ float tile[32][33];
  int k0 = blockIdx.x * 32, n0 = blockIdx.y * 32;
  int tx = threadIdx.x, ty = threadIdx.y;   // block (32, 8)
  #pragma unroll
  for (int i = ty; i < 32; i += 8) {
    int k = k0 + i, n = n0 + tx;
    tile[i][tx] = (k < K && n < N) ? W[(size_t)k * N + n] : 0.f;
  }
  __syncthreads();
  #pragma unroll
  for (int i = ty; i < 32; i += 8) {
    int n = n0 + i, k = k0 + tx;
    if (n < N && k < K) WT[(size_t)n * K + k] = f2bf(tile[tx][i]);
  }
}

// ---------------- WMMA GEMM ----------------
// out[M,N] = act[M,K](bf16) @ W[K,N]  using WT = W^T [N,K] bf16.
// Requires M%64==0, N%256==0, K%64==0.
// Block tile 64x256; 8 waves, each wave computes 32x64 (2x4 WMMA tiles).
constexpr int MBLK  = 64;
constexpr int NBLK  = 256;
constexpr int KSTEP = 64;
constexpr int LDT   = KSTEP + 8;   // 72: keeps 16B alignment (144B rows)

template <int ACT, bool HAS_RES, bool OUT_F32, bool OUT_BF16>
__global__ __launch_bounds__(256)
void gemm_bf16_kernel(const __bf16* __restrict__ A,   // [M,K]
                      const __bf16* __restrict__ WT,  // [N,K]
                      const float*  __restrict__ bias,// [N]
                      const float*  __restrict__ resid,// [M,N]
                      float*        __restrict__ outf, // [M,N]
                      __bf16*       __restrict__ outb, // [M,N]
                      int Md, int Nd, int Kd) {
  __shared__ __bf16 As[MBLK][LDT];   //  9.2 KB
  __shared__ __bf16 Bs[NBLK][LDT];   // 36.9 KB

  const int m0 = blockIdx.y * MBLK;
  const int n0 = blockIdx.x * NBLK;
  const int t  = threadIdx.x;
  const int wave = __builtin_amdgcn_readfirstlane(t >> 5);  // scalar, uniform
  const int wm = wave >> 2;   // 0..1 (32 rows each)
  const int wn = wave & 3;    // 0..3 (64 cols each)

  v8f acc[2][4];
  #pragma unroll
  for (int i = 0; i < 2; ++i)
    #pragma unroll
    for (int j = 0; j < 4; ++j)
      acc[i][j] = v8f{0.f, 0.f, 0.f, 0.f, 0.f, 0.f, 0.f, 0.f};

  for (int k0 = 0; k0 < Kd; k0 += KSTEP) {
    // stage A tile: 64 x 64 bf16 (16B per ld, 2 iters)
    #pragma unroll
    for (int i = t; i < (MBLK * KSTEP / 8); i += 256) {
      int r  = i >> 3;
      int cc = (i & 7) * 8;
      *reinterpret_cast<uint4*>(&As[r][cc]) =
          *reinterpret_cast<const uint4*>(&A[(size_t)(m0 + r) * Kd + k0 + cc]);
    }
    // stage W^T tile: 256 x 64 bf16 (8 iters)
    #pragma unroll
    for (int i = t; i < (NBLK * KSTEP / 8); i += 256) {
      int r  = i >> 3;
      int cc = (i & 7) * 8;
      *reinterpret_cast<uint4*>(&Bs[r][cc]) =
          *reinterpret_cast<const uint4*>(&WT[(size_t)(n0 + r) * Kd + k0 + cc]);
    }
    __syncthreads();

    #pragma unroll
    for (int kk = 0; kk < KSTEP; kk += 32) {
      v16bf af[2], bfq[4];
      #pragma unroll
      for (int mi = 0; mi < 2; ++mi)
        af[mi] = load_frag_A(&As[wm * 32 + mi * 16][kk], LDT);
      #pragma unroll
      for (int ni = 0; ni < 4; ++ni)
        bfq[ni] = load_frag_B(&Bs[wn * 64 + ni * 16][kk], LDT);
      #pragma unroll
      for (int mi = 0; mi < 2; ++mi)
        #pragma unroll
        for (int ni = 0; ni < 4; ++ni)
          acc[mi][ni] = wmma_bf16(af[mi], bfq[ni], acc[mi][ni]);
    }
    __syncthreads();
  }

  const int lane = t & 31;
  const int lm = (lane >> 4) * 8;
  const int lc = lane & 15;
  #pragma unroll
  for (int mi = 0; mi < 2; ++mi) {
    #pragma unroll
    for (int ni = 0; ni < 4; ++ni) {
      int gm0 = m0 + wm * 32 + mi * 16;
      int gn  = n0 + wn * 64 + ni * 16 + lc;
      float bcol = bias ? bias[gn] : 0.f;
      #pragma unroll
      for (int v = 0; v < 8; ++v) {
        int gm = gm0 + v + lm;
        float val = acc[mi][ni][v] + bcol;
        if (ACT == 1) val = gelu_exact(val);
        if (HAS_RES) val += resid[(size_t)gm * Nd + gn];
        if (OUT_F32)  outf[(size_t)gm * Nd + gn] = val;
        if (OUT_BF16) outb[(size_t)gm * Nd + gn] = f2bf(val);
      }
    }
  }
}

// ---------------- fused attention ----------------
// grid: (S/32, NH, B); block 256 (8 waves).
// Q/K/V bf16 [M, H]; per-head slice at col h*64. Output ctx bf16 [M, H].
constexpr int A_LDK = DHh + 8;       // 72
constexpr int A_LDV = 128 + 8;       // 136
constexpr int A_LDS = Ss + 4;        // 516 (fp32 scores)
constexpr int A_LDP = Ss + 8;        // 520 (bf16 probs)
constexpr int ATT_SMEM =
    32 * A_LDK * 2 +                 // Qs
    128 * A_LDK * 2 +                // Ks / VsT (reused)
    32 * A_LDS * 4 +                 // scores fp32
    32 * A_LDP * 2;                  // probs bf16  (~120 KB, CDNA5 LDS is 320 KB)

__global__ __launch_bounds__(256)
void attention_kernel(const __bf16* __restrict__ Qb, const __bf16* __restrict__ Kb,
                      const __bf16* __restrict__ Vb, const int* __restrict__ amask,
                      __bf16* __restrict__ ctxb) {
  extern __shared__ char smem[];
  __bf16* Qs  = reinterpret_cast<__bf16*>(smem);
  __bf16* Ks  = Qs + 32 * A_LDK;
  __bf16* VsT = Ks;  // reuse K region in pass 2 (64*136*2 <= 128*72*2)
  float*  Sc  = reinterpret_cast<float*>(Ks + 128 * A_LDK);
  __bf16* Ps  = reinterpret_cast<__bf16*>(Sc + 32 * A_LDS);

  const int qb = blockIdx.x;   // 0..15
  const int h  = blockIdx.y;
  const int b  = blockIdx.z;
  const int q0 = qb * 32;
  const int t  = threadIdx.x;
  const int wave = __builtin_amdgcn_readfirstlane(t >> 5);  // scalar, uniform
  const int lane = t & 31;
  const int lm = (lane >> 4) * 8;
  const int lc = lane & 15;

  // stage Q: 32 x 64 bf16 (one 16B load per thread)
  {
    int r  = t >> 3;
    int cc = (t & 7) * 8;
    *reinterpret_cast<uint4*>(&Qs[r * A_LDK + cc]) =
        *reinterpret_cast<const uint4*>(
            &Qb[(size_t)(b * Ss + q0 + r) * Hh + h * DHh + cc]);
  }
  __syncthreads();

  // ---- pass 1: scores = (Q K^T) * scale + mask ----
  for (int c = 0; c < 4; ++c) {
    #pragma unroll
    for (int i = t; i < (128 * DHh / 8); i += 256) {
      int r  = i >> 3;
      int cc = (i & 7) * 8;
      *reinterpret_cast<uint4*>(&Ks[r * A_LDK + cc]) =
          *reinterpret_cast<const uint4*>(
              &Kb[(size_t)(b * Ss + c * 128 + r) * Hh + h * DHh + cc]);
    }
    __syncthreads();

    // two 16x16 score tiles per wave, fully unrolled (wave-uniform control flow)
    #pragma unroll
    for (int rep = 0; rep < 2; ++rep) {
      const int tix = wave + rep * 8;   // 0..15
      const int tm = tix >> 3;          // 0..1
      const int tn = tix & 7;           // 0..7
      v8f s = v8f{0.f, 0.f, 0.f, 0.f, 0.f, 0.f, 0.f, 0.f};
      #pragma unroll
      for (int kk = 0; kk < DHh; kk += 32) {
        v16bf aq = load_frag_A(&Qs[(tm * 16) * A_LDK + kk], A_LDK);
        v16bf bk = load_frag_B(&Ks[(tn * 16) * A_LDK + kk], A_LDK);
        s = wmma_bf16(aq, bk, s);
      }
      int jcol = c * 128 + tn * 16 + lc;
      float mterm = (1.0f - (float)amask[b * Ss + jcol]) * -1e9f;
      #pragma unroll
      for (int v = 0; v < 8; ++v)
        Sc[(tm * 16 + v + lm) * A_LDS + jcol] = s[v] * 0.125f + mterm;
    }
    __syncthreads();
  }

  // ---- softmax over rows of Sc, write bf16 probs to Ps ----
  #pragma unroll
  for (int rr = 0; rr < 4; ++rr) {
    int r = wave * 4 + rr;
    float* row = &Sc[r * A_LDS];
    float mx = -1e30f;
    for (int j = lane; j < Ss; j += 32) mx = fmaxf(mx, row[j]);
    mx = wave_max(mx);
    float sum = 0.f;
    for (int j = lane; j < Ss; j += 32) {
      float e = expf(row[j] - mx);
      row[j] = e;
      sum += e;
    }
    sum = wave_sum(sum);
    float inv = 1.0f / sum;
    for (int j = lane; j < Ss; j += 32) Ps[r * A_LDP + j] = f2bf(row[j] * inv);
  }
  __syncthreads();

  // ---- pass 2: ctx = P V ---- (each wave owns one 16x16 output tile)
  const int tm = wave >> 2;   // 0..1
  const int tn = wave & 3;    // 0..3
  v8f ctx = v8f{0.f, 0.f, 0.f, 0.f, 0.f, 0.f, 0.f, 0.f};
  for (int c = 0; c < 4; ++c) {
    for (int i = t; i < DHh * 128; i += 256) {
      int d  = i & 63;
      int kv = i >> 6;
      VsT[d * A_LDV + kv] =
          Vb[(size_t)(b * Ss + c * 128 + kv) * Hh + h * DHh + d];
    }
    __syncthreads();
    #pragma unroll
    for (int kk = 0; kk < 128; kk += 32) {
      v16bf ap = load_frag_A(&Ps[(tm * 16) * A_LDP + c * 128 + kk], A_LDP);
      v16bf bv = load_frag_B(&VsT[(tn * 16) * A_LDV + kk], A_LDV);
      ctx = wmma_bf16(ap, bv, ctx);
    }
    __syncthreads();
  }
  {
    int gq = b * Ss + q0 + tm * 16;
    int gc = h * DHh + tn * 16 + lc;
    #pragma unroll
    for (int v = 0; v < 8; ++v)
      ctxb[(size_t)(gq + v + lm) * Hh + gc] = f2bf(ctx[v]);
  }
}

// ---------------- LayerNorm (optionally + residual), optional bf16 copy ----------------
__global__ __launch_bounds__(256)
void layernorm_kernel(const float* __restrict__ a, const float* __restrict__ resid,
                      const float* __restrict__ g, const float* __restrict__ bb,
                      float* __restrict__ outf, __bf16* __restrict__ outb) {
  const int m = blockIdx.x;
  const int t = threadIdx.x;
  const int wave = t >> 5, lane = t & 31;
  __shared__ float rs[8], rs2[8];

  float s = 0.f, s2 = 0.f;
  for (int j = t; j < Hh; j += 256) {
    float v = a[(size_t)m * Hh + j];
    if (resid) v += resid[(size_t)m * Hh + j];
    s += v;
    s2 += v * v;
  }
  s = wave_sum(s);
  s2 = wave_sum(s2);
  if (lane == 0) { rs[wave] = s; rs2[wave] = s2; }
  __syncthreads();
  float S = 0.f, S2 = 0.f;
  #pragma unroll
  for (int i = 0; i < 8; ++i) { S += rs[i]; S2 += rs2[i]; }
  float mean = S / (float)Hh;
  float var  = S2 / (float)Hh - mean * mean;
  float r    = rsqrtf(var + 1e-12f);
  for (int j = t; j < Hh; j += 256) {
    float v = a[(size_t)m * Hh + j];
    if (resid) v += resid[(size_t)m * Hh + j];
    float y = (v - mean) * r * g[j] + bb[j];
    outf[(size_t)m * Hh + j] = y;
    if (outb) outb[(size_t)m * Hh + j] = f2bf(y);
  }
}

// ---------------- gating: logits = x@Wg + bg; top-2; softmax ----------------
__global__ __launch_bounds__(256)
void gate_topk_kernel(const float* __restrict__ x, const float* __restrict__ Wg,
                      const float* __restrict__ bg, float* __restrict__ score,
                      int* __restrict__ idx) {
  const int wave = threadIdx.x >> 5, lane = threadIdx.x & 31;
  const int m = blockIdx.x * 8 + wave;
  float l[Ee];
  #pragma unroll
  for (int e = 0; e < Ee; ++e) l[e] = 0.f;
  for (int hh = lane; hh < Hh; hh += 32) {
    float xv = x[(size_t)m * Hh + hh];
    #pragma unroll
    for (int e = 0; e < Ee; ++e) l[e] += xv * Wg[hh * Ee + e];
  }
  #pragma unroll
  for (int e = 0; e < Ee; ++e) l[e] = wave_sum(l[e]);
  if (lane == 0) {
    #pragma unroll
    for (int e = 0; e < Ee; ++e) l[e] += bg[e];
    int e0 = 0; float v0 = l[0];
    #pragma unroll
    for (int e = 1; e < Ee; ++e) if (l[e] > v0) { v0 = l[e]; e0 = e; }
    int e1 = -1; float v1 = -1e30f;
    #pragma unroll
    for (int e = 0; e < Ee; ++e)
      if (e != e0 && l[e] > v1) { v1 = l[e]; e1 = e; }
    float e10 = expf(v1 - v0);
    float p0 = 1.0f / (1.0f + e10);
    float p1 = e10 / (1.0f + e10);
    score[m * 2 + 0] = p0; score[m * 2 + 1] = p1;
    idx[m * 2 + 0] = e0;   idx[m * 2 + 1] = e1;
  }
}

// ---------------- adapter combine: ACC += SCALING * sum_j s_j*(h_ej @ A2[ej] + a2b[ej]) --------
__global__ __launch_bounds__(256)
void adapter_combine_kernel(const float* __restrict__ her, const float* __restrict__ A2,
                            const float* __restrict__ a2b, const float* __restrict__ score,
                            const int* __restrict__ idx, float* __restrict__ acc) {
  const int hcol = blockIdx.x * 256 + threadIdx.x;   // < 768
  const int m = blockIdx.y;
  float sum = 0.f;
  #pragma unroll
  for (int j = 0; j < 2; ++j) {
    int e = idx[m * 2 + j];
    float sc = score[m * 2 + j];
    float a = a2b[e * Hh + hcol];
    const float* hp = &her[(size_t)m * ERr + e * Rr];
    const float* ap = &A2[(size_t)e * Rr * Hh + hcol];
    #pragma unroll 8
    for (int r = 0; r < Rr; ++r) a += hp[r] * ap[(size_t)r * Hh];
    sum += sc * a;
  }
  acc[(size_t)m * Hh + hcol] += 2.0f * sum;   // SCALING = 2.0
}

// ---------------- host launcher ----------------
extern "C" void kernel_launch(void* const* d_in, const int* in_sizes, int n_in,
                              void* d_out, int out_size, void* d_ws, size_t ws_size,
                              hipStream_t stream) {
  (void)in_sizes; (void)n_in; (void)out_size; (void)ws_size;
  const float* hid   = (const float*)d_in[0];
  const int*   amask = (const int*)  d_in[1];
  const float* Wq = (const float*)d_in[2],  *bq  = (const float*)d_in[3];
  const float* Wk = (const float*)d_in[4],  *bk  = (const float*)d_in[5];
  const float* Wv = (const float*)d_in[6],  *bv  = (const float*)d_in[7];
  const float* Wao= (const float*)d_in[8],  *bao = (const float*)d_in[9];
  const float* g1 = (const float*)d_in[10], *be1 = (const float*)d_in[11];
  const float* Wi = (const float*)d_in[12], *bi  = (const float*)d_in[13];
  const float* Wo = (const float*)d_in[14], *bo  = (const float*)d_in[15];
  const float* g2 = (const float*)d_in[16], *be2 = (const float*)d_in[17];
  const float* A1 = (const float*)d_in[18], *a1b = (const float*)d_in[19];
  const float* A2 = (const float*)d_in[20], *a2b = (const float*)d_in[21];
  const float* Wg = (const float*)d_in[22], *bg  = (const float*)d_in[23];

  // ---- carve workspace ----
  char* p = (char*)d_ws;
  auto carve = [&](size_t bytes) -> char* {
    char* r = p;
    p += (bytes + 255) & ~(size_t)255;
    return r;
  };
  float*  X     = (float*) carve((size_t)Mm * Hh * 4);
  float*  ATT   = (float*) carve((size_t)Mm * Hh * 4);
  float*  ACC   = (float*) carve((size_t)Mm * Hh * 4);
  float*  HER   = (float*) carve((size_t)Mm * ERr * 4);
  float*  SCORE = (float*) carve((size_t)Mm * 2 * 4);
  int*    IDX   = (int*)   carve((size_t)Mm * 2 * 4);
  __bf16* Xb    = (__bf16*)carve((size_t)Mm * Hh * 2);
  __bf16* ATTb  = (__bf16*)carve((size_t)Mm * Hh * 2);
  __bf16* Qb    = (__bf16*)carve((size_t)Mm * Hh * 2);
  __bf16* Kb    = (__bf16*)carve((size_t)Mm * Hh * 2);
  __bf16* Vb    = (__bf16*)carve((size_t)Mm * Hh * 2);
  __bf16* CTXb  = (__bf16*)carve((size_t)Mm * Hh * 2);
  __bf16* FFb   = (__bf16*)carve((size_t)Mm * FFf * 2);
  __bf16* WT    = (__bf16*)carve((size_t)Hh * FFf * 2);

  const dim3 tb32x8(32, 8);
  const dim3 gemmHH(Hh / NBLK, Mm / MBLK);    // (3, 64)
  const dim3 gemmHF(FFf / NBLK, Mm / MBLK);   // (12, 64)
  const dim3 gemmHE(ERr / NBLK, Mm / MBLK);   // (2, 64)

  for (int l = 0; l < NLl; ++l) {
    const float* Xsrc = (l == 0) ? hid : X;
    float* Xdst = (l == NLl - 1) ? (float*)d_out : X;

    // activations -> bf16
    cvt_bf16_kernel<<<(Mm * Hh + 255) / 256, 256, 0, stream>>>(Xsrc, Xb, Mm * Hh);

    // Q
    transpose_cvt_kernel<<<dim3(Hh / 32, Hh / 32), tb32x8, 0, stream>>>(
        Wq + (size_t)l * Hh * Hh, WT, Hh, Hh);
    gemm_bf16_kernel<0, false, false, true><<<gemmHH, 256, 0, stream>>>(
        Xb, WT, bq + l * Hh, nullptr, nullptr, Qb, Mm, Hh, Hh);
    // K
    transpose_cvt_kernel<<<dim3(Hh / 32, Hh / 32), tb32x8, 0, stream>>>(
        Wk + (size_t)l * Hh * Hh, WT, Hh, Hh);
    gemm_bf16_kernel<0, false, false, true><<<gemmHH, 256, 0, stream>>>(
        Xb, WT, bk + l * Hh, nullptr, nullptr, Kb, Mm, Hh, Hh);
    // V
    transpose_cvt_kernel<<<dim3(Hh / 32, Hh / 32), tb32x8, 0, stream>>>(
        Wv + (size_t)l * Hh * Hh, WT, Hh, Hh);
    gemm_bf16_kernel<0, false, false, true><<<gemmHH, 256, 0, stream>>>(
        Xb, WT, bv + l * Hh, nullptr, nullptr, Vb, Mm, Hh, Hh);

    // fused attention -> CTXb
    attention_kernel<<<dim3(Ss / 32, NHh, Bb), 256, ATT_SMEM, stream>>>(
        Qb, Kb, Vb, amask, CTXb);

    // attention output projection + residual -> ACC (fp32)
    transpose_cvt_kernel<<<dim3(Hh / 32, Hh / 32), tb32x8, 0, stream>>>(
        Wao + (size_t)l * Hh * Hh, WT, Hh, Hh);
    gemm_bf16_kernel<0, true, true, false><<<gemmHH, 256, 0, stream>>>(
        CTXb, WT, bao + l * Hh, Xsrc, ACC, nullptr, Mm, Hh, Hh);

    // LN1 -> ATT (fp32 + bf16)
    layernorm_kernel<<<Mm, 256, 0, stream>>>(ACC, nullptr, g1 + l * Hh, be1 + l * Hh,
                                             ATT, ATTb);

    // FFN up (GELU) -> FFb bf16
    transpose_cvt_kernel<<<dim3(Hh / 32, FFf / 32), tb32x8, 0, stream>>>(
        Wi + (size_t)l * Hh * FFf, WT, Hh, FFf);
    gemm_bf16_kernel<1, false, false, true><<<gemmHF, 256, 0, stream>>>(
        ATTb, WT, bi + l * FFf, nullptr, nullptr, FFb, Mm, FFf, Hh);

    // FFN down -> ACC fp32
    transpose_cvt_kernel<<<dim3(FFf / 32, Hh / 32), tb32x8, 0, stream>>>(
        Wo + (size_t)l * FFf * Hh, WT, FFf, Hh);
    gemm_bf16_kernel<0, false, true, false><<<gemmHH, 256, 0, stream>>>(
        FFb, WT, bo + l * Hh, nullptr, ACC, nullptr, Mm, Hh, FFf);

    // adapter up-proj (GELU) -> HER fp32
    transpose_cvt_kernel<<<dim3(Hh / 32, ERr / 32), tb32x8, 0, stream>>>(
        A1 + (size_t)l * Hh * ERr, WT, Hh, ERr);
    gemm_bf16_kernel<1, false, true, false><<<gemmHE, 256, 0, stream>>>(
        ATTb, WT, a1b + l * ERr, nullptr, HER, nullptr, Mm, ERr, Hh);

    // gate -> top-2 scores/indices
    gate_topk_kernel<<<Mm / 8, 256, 0, stream>>>(ATT, Wg + (size_t)l * Hh * Ee,
                                                 bg + l * Ee, SCORE, IDX);

    // sparse expert down-proj, accumulate into ACC
    adapter_combine_kernel<<<dim3(Hh / 256, Mm), 256, 0, stream>>>(
        HER, A2 + (size_t)l * Ee * Rr * Hh, a2b + (size_t)l * Ee * Hh, SCORE, IDX, ACC);

    // LN2 over (ffn + adapter + attn-residual) -> next X (or d_out)
    layernorm_kernel<<<Mm, 256, 0, stream>>>(ACC, ATT, g2 + l * Hh, be2 + l * Hh,
                                             Xdst, nullptr);
  }
}
